// MultiHeadGraphAttention_75084618269416
// MI455X (gfx1250) — matrix operand order repaired and checked by reading, not compiled, not run on hardware
//
#include <hip/hip_runtime.h>
#include <hip/hip_bf16.h>

// ---------------------------------------------------------------------------
// MultiHeadGraphAttention for MI455X (gfx1250), wave32 + WMMA bf16 16x16x32.
// Flash-style fused softmax(alpha) @ h to avoid 537MB of alpha traffic.
// ---------------------------------------------------------------------------

typedef __bf16 bf16_t;
typedef __attribute__((ext_vector_type(16))) __bf16 v16bf;
typedef __attribute__((ext_vector_type(8)))  __bf16 v8bf;
typedef __attribute__((ext_vector_type(8)))  float  v8f;

#define B_   8
#define N_   2048
#define DIN  128
#define H_   4
#define DH   64
#define E_   256           // H_*DH
#define NEG_SLOPE 0.2f

// Workspace layout (bytes). Total ~17.4 MB.
#define OFF_PROJW_BF  (size_t)0          // 256*128 bf16  = 65536
#define OFF_OUTW_BF   (size_t)65536      // 64*256  bf16  = 32768
#define OFF_HT_BF     (size_t)98304      // B*H*DH*N bf16 = 8388608  (transposed h)
#define OFF_SRC       (size_t)8486912    // B*H*N f32     = 262144
#define OFF_DST       (size_t)8749056    // B*H*N f32     = 262144
#define OFF_OUT_BF    (size_t)9011200    // B*N*E bf16    = 8388608
// end: 17399808 bytes

// -------------------------- weight conversion ------------------------------
__global__ void convert_kernel(const float* __restrict__ pw,
                               const float* __restrict__ ow,
                               bf16_t* __restrict__ pwb,
                               bf16_t* __restrict__ owb) {
  int t = blockIdx.x * blockDim.x + threadIdx.x;
  if (t < E_ * DIN) pwb[t] = (bf16_t)pw[t];
  if (t < DH * E_)  owb[t] = (bf16_t)ow[t];
}

// ------------------------- projection: h = x @ W^T -------------------------
// Wave computes a 16-row x 64-col tile (one head). Block = 4 waves = all 4 heads.
// Writes h transposed: hT[(b*H+head)*DH + dh][n]  (bf16) for fast B-operand loads.
__global__ __launch_bounds__(128)
void proj_kernel(const float* __restrict__ x,
                 const bf16_t* __restrict__ pw,
                 bf16_t* __restrict__ hT) {
  const int lane = threadIdx.x & 31;
  const int wave = threadIdx.x >> 5;      // = head
  const int bIdx = blockIdx.x >> 7;       // 128 row-tiles per batch
  const int rowBase = (blockIdx.x & 127) * 16;
  const int hi   = lane >> 4;
  const int col  = lane & 15;
  const int klo  = hi ? 8 : 0;            // A-operand K base for this lane half
  const int koff = hi ? 16 : 0;           // unused; B operand also uses klo
  const int head = wave;

  v8f acc[4] = {v8f{}, v8f{}, v8f{}, v8f{}};
  const float* xrow = x + ((size_t)bIdx * N_ + rowBase + col) * DIN;

  for (int kk = 0; kk < DIN; kk += 32) {
    v16bf a;
    #pragma unroll
    for (int g = 0; g < 2; ++g) {         // K = kk+klo+0..7 and kk+16+klo+0..7
      const float4* xp = (const float4*)(xrow + kk + g * 16 + klo);
      float4 f0 = xp[0], f1 = xp[1];
      a[g*8+0]=(bf16_t)f0.x; a[g*8+1]=(bf16_t)f0.y; a[g*8+2]=(bf16_t)f0.z; a[g*8+3]=(bf16_t)f0.w;
      a[g*8+4]=(bf16_t)f1.x; a[g*8+5]=(bf16_t)f1.y; a[g*8+6]=(bf16_t)f1.z; a[g*8+7]=(bf16_t)f1.w;
    }
    #pragma unroll
    for (int sub = 0; sub < 4; ++sub) {
      const int e = head * 64 + sub * 16 + col;      // output column (B-operand N)
      const bf16_t* bp = pw + (size_t)e * DIN + kk + klo;
      v8bf b0 = *(const v8bf*)bp, b1 = *(const v8bf*)(bp + 16);
      v16bf bm = __builtin_shufflevector(b0, b1, 0,1,2,3,4,5,6,7,8,9,10,11,12,13,14,15);
      acc[sub] = __builtin_amdgcn_wmma_f32_16x16x32_bf16(
          false, a, false, bm, (short)0, acc[sub], false, false);
    }
  }
  // C layout: vgpr r -> row r + 8*hi, col = lane&15. 8 consecutive n per lane.
  #pragma unroll
  for (int sub = 0; sub < 4; ++sub) {
    const int dh = sub * 16 + col;
    v8bf o;
    #pragma unroll
    for (int r = 0; r < 8; ++r) o[r] = (bf16_t)acc[sub][r];
    bf16_t* dst = hT + (((size_t)bIdx * H_ + head) * DH + dh) * N_ + rowBase + hi * 8;
    *(v8bf*)dst = o;
  }
}

// -------------------- src/dst attention coefficient dots -------------------
__global__ void srcdst_kernel(const bf16_t* __restrict__ hT,
                              const float* __restrict__ asrc,
                              const float* __restrict__ adst,
                              float* __restrict__ srcv,
                              float* __restrict__ dstv) {
  int t = blockIdx.x * blockDim.x + threadIdx.x;   // over B*H*N
  if (t >= B_ * H_ * N_) return;
  int n = t % N_;
  int bh = t / N_;
  int head = bh % H_;
  const bf16_t* hp = hT + (size_t)bh * DH * N_ + n;
  float s = 0.f, d = 0.f;
  #pragma unroll 8
  for (int dh = 0; dh < DH; ++dh) {
    float hv = (float)hp[(size_t)dh * N_];
    s += hv * asrc[head * DH + dh];
    d += hv * adst[head * DH + dh];
  }
  srcv[t] = s;
  dstv[t] = d;
}

// ------------------- fused masked-softmax attention @ h --------------------
// Wave owns one (b, head, 16-row i-tile). Streams j in 32-wide tiles:
// builds p-tile directly in the WMMA A-operand lane layout, does online
// softmax (shfl_xor across the two lane halves), accumulates p @ h via WMMA.
__global__ __launch_bounds__(128)
void attn_kernel(const float* __restrict__ adj,
                 const bf16_t* __restrict__ hT,
                 const float* __restrict__ srcv,
                 const float* __restrict__ dstv,
                 bf16_t* __restrict__ outbf) {
  const int lane = threadIdx.x & 31;
  const int wave = threadIdx.x >> 5;
  const int wid  = blockIdx.x * 4 + wave;          // 0..4095
  const int bIdx = wid >> 9;                       // / (H_*128)
  const int head = (wid >> 7) & (H_ - 1);
  const int iBase = (wid & 127) * 16;
  const int hi = lane >> 4;
  const int col = lane & 15;
  const int klo = hi ? 8 : 0;
  const int koff = hi ? 16 : 0;
  const int bh = bIdx * H_ + head;

  const float  s_i  = srcv[(size_t)bh * N_ + iBase + col];
  const float* mrow = adj  + (size_t)(iBase + col) * N_;
  const float* drow = dstv + (size_t)bh * N_;

  float m = -1e30f, l = 0.f;
  v8f acc[4] = {v8f{}, v8f{}, v8f{}, v8f{}};

  for (int jBase = 0; jBase < N_; jBase += 32) {
    if (jBase + 32 < N_) __builtin_prefetch(mrow + jBase + 32, 0, 0);

    float ef[16];
    #pragma unroll
    for (int g = 0; g < 2; ++g) {
      const int j0 = jBase + g * 16 + klo;
      const float4* mp = (const float4*)(mrow + j0);
      const float4* dp = (const float4*)(drow + j0);
      float4 m0 = mp[0], m1 = mp[1];
      float4 d0 = dp[0], d1 = dp[1];
      float dv[8] = {d0.x, d0.y, d0.z, d0.w, d1.x, d1.y, d1.z, d1.w};
      float mv[8] = {m0.x, m0.y, m0.z, m0.w, m1.x, m1.y, m1.z, m1.w};
      #pragma unroll
      for (int i = 0; i < 8; ++i) {
        float e = s_i + dv[i];
        e = (e >= 0.f) ? e : NEG_SLOPE * e;
        ef[g * 8 + i] = (mv[i] < 0.5f) ? -1.0e9f : e;
      }
    }
    // online softmax: this lane holds 16 of row (lane&15)'s 32 logits
    float tmax = ef[0];
    #pragma unroll
    for (int i = 1; i < 16; ++i) tmax = fmaxf(tmax, ef[i]);
    tmax = fmaxf(tmax, __shfl_xor(tmax, 16, 32));
    const float mnew = fmaxf(m, tmax);
    const float fac  = __expf(m - mnew);
    float rsum = 0.f;
    v16bf a;
    #pragma unroll
    for (int i = 0; i < 16; ++i) {
      float p = __expf(ef[i] - mnew);
      rsum += p;
      a[i] = (bf16_t)p;
    }
    rsum += __shfl_xor(rsum, 16, 32);
    l = l * fac + rsum;
    m = mnew;

    float facr[8];
    #pragma unroll
    for (int r = 0; r < 8; ++r) facr[r] = __shfl(fac, r + hi * 8, 32);

    #pragma unroll
    for (int sub = 0; sub < 4; ++sub) {
      v8f c = acc[sub];
      #pragma unroll
      for (int r = 0; r < 8; ++r) c[r] *= facr[r];
      const bf16_t* hp = hT + ((size_t)bh * DH + sub * 16 + col) * N_ + jBase + klo;
      v8bf b0 = *(const v8bf*)hp, b1 = *(const v8bf*)(hp + 16);
      v16bf bm = __builtin_shufflevector(b0, b1, 0,1,2,3,4,5,6,7,8,9,10,11,12,13,14,15);
      acc[sub] = __builtin_amdgcn_wmma_f32_16x16x32_bf16(
          false, a, false, bm, (short)0, c, false, false);
    }
  }

  const float rl = 1.0f / l;
  float rlr[8];
  #pragma unroll
  for (int r = 0; r < 8; ++r) rlr[r] = __shfl(rl, r + hi * 8, 32);
  #pragma unroll
  for (int sub = 0; sub < 4; ++sub) {
    #pragma unroll
    for (int r = 0; r < 8; ++r) {
      const int n = iBase + hi * 8 + r;
      const int e = head * 64 + sub * 16 + col;     // transpose(0,2,1,3) fold
      outbf[((size_t)bIdx * N_ + n) * E_ + e] = (bf16_t)(acc[sub][r] * rlr[r]);
    }
  }
}

// --------------------- final projection: out @ out_w^T + b -----------------
__global__ __launch_bounds__(128)
void final_kernel(const bf16_t* __restrict__ outbf,
                  const bf16_t* __restrict__ oww,
                  const float* __restrict__ outb,
                  float* __restrict__ y) {
  const int lane = threadIdx.x & 31;
  const int wave = threadIdx.x >> 5;
  const int rowBase = (blockIdx.x * 4 + wave) * 16;   // over B*N flattened rows
  const int hi = lane >> 4;
  const int col = lane & 15;
  const int klo = hi ? 8 : 0;
  const int koff = hi ? 16 : 0;

  v8f acc[4] = {v8f{}, v8f{}, v8f{}, v8f{}};
  const bf16_t* arow = outbf + (size_t)(rowBase + col) * E_;

  for (int kk = 0; kk < E_; kk += 32) {
    v16bf a;
    #pragma unroll
    for (int g = 0; g < 2; ++g) {
      union { uint4 u; bf16_t h[8]; } t;
      t.u = *(const uint4*)(arow + kk + g * 16 + klo);
      #pragma unroll
      for (int i = 0; i < 8; ++i) a[g * 8 + i] = t.h[i];
    }
    #pragma unroll
    for (int sub = 0; sub < 4; ++sub) {
      const int o = sub * 16 + col;
      const bf16_t* bp = oww + (size_t)o * E_ + kk + klo;
      v8bf b0 = *(const v8bf*)bp, b1 = *(const v8bf*)(bp + 16);
      v16bf bm = __builtin_shufflevector(b0, b1, 0,1,2,3,4,5,6,7,8,9,10,11,12,13,14,15);
      acc[sub] = __builtin_amdgcn_wmma_f32_16x16x32_bf16(
          false, a, false, bm, (short)0, acc[sub], false, false);
    }
  }
  #pragma unroll
  for (int sub = 0; sub < 4; ++sub) {
    const int o = sub * 16 + col;
    const float bias = outb[o];
    #pragma unroll
    for (int r = 0; r < 8; ++r) {
      const int row = rowBase + hi * 8 + r;
      y[(size_t)row * DH + o] = acc[sub][r] + bias;
    }
  }
}

// ------------------------------- launcher ----------------------------------
extern "C" void kernel_launch(void* const* d_in, const int* in_sizes, int n_in,
                              void* d_out, int out_size, void* d_ws, size_t ws_size,
                              hipStream_t stream) {
  const float* x        = (const float*)d_in[0];   // (8,2048,128)
  const float* adj_mask = (const float*)d_in[1];   // (2048,2048)
  const float* proj_w   = (const float*)d_in[2];   // (256,128)
  const float* attn_src = (const float*)d_in[3];   // (4,64)
  const float* attn_dst = (const float*)d_in[4];   // (4,64)
  const float* out_w    = (const float*)d_in[5];   // (64,256)
  const float* out_b    = (const float*)d_in[6];   // (64,)
  float* y = (float*)d_out;                        // (8,2048,64)

  char* ws = (char*)d_ws;                          // needs ~17.4 MB
  bf16_t* pwb   = (bf16_t*)(ws + OFF_PROJW_BF);
  bf16_t* owb   = (bf16_t*)(ws + OFF_OUTW_BF);
  bf16_t* hT    = (bf16_t*)(ws + OFF_HT_BF);
  float*  srcv  = (float*) (ws + OFF_SRC);
  float*  dstv  = (float*) (ws + OFF_DST);
  bf16_t* outbf = (bf16_t*)(ws + OFF_OUT_BF);

  convert_kernel<<<(E_ * DIN + 255) / 256, 256, 0, stream>>>(proj_w, out_w, pwb, owb);
  proj_kernel<<<B_ * (N_ / 16), 128, 0, stream>>>(x, pwb, hT);
  srcdst_kernel<<<(B_ * H_ * N_) / 256, 256, 0, stream>>>(hT, attn_src, attn_dst, srcv, dstv);
  attn_kernel<<<(B_ * H_ * (N_ / 16)) / 4, 128, 0, stream>>>(adj_mask, hT, srcv, dstv, outbf);
  final_kernel<<<(B_ * N_ / 16) / 4, 128, 0, stream>>>(outbf, owb, out_b, y);
}
